// PureTriadicBrain_73753178407497
// MI455X (gfx1250) — matrix operand order repaired and checked by reading, not compiled, not run on hardware
//
#include <hip/hip_runtime.h>
#include <hip/hip_bf16.h>
#include <stdint.h>

typedef __attribute__((ext_vector_type(16))) _Float16 v16h;
typedef __attribute__((ext_vector_type(8)))  float    v8f;
typedef __attribute__((ext_vector_type(4)))  _Float16 h4_;

#define B_   4
#define N_   128
#define D_   256
#define BN_  (B_*N_)            // 512
#define BND_ (BN_*D_)           // 131072
#define BNN_ (B_*N_*N_)         // 65536
#define WSZ_ ((size_t)BN_*D_*D_)// 33554432 per weight tensor

#define EMA_SPEED_   0.05f
#define LR_          0.024f
#define MOMENTUM_    0.4f
#define UNTANH_S_    0.6f
#define UNTANH_M_    1.8477590650225735f   // sqrt(2+sqrt(2))
#define SPEED_       0.5f
#define WIDTH_       0.01f
#define WD_          0.01f
#define NOISE_SCALE_ 0.8f
#define C_MOMENT_    0.21f
#define W_SCALE_     1.0f
#define A_VAR_       0.05f
#define EPS_         1e-8f

#if defined(__has_builtin)
#if __has_builtin(__builtin_amdgcn_tensor_load_to_lds) && __has_builtin(__builtin_amdgcn_s_wait_tensorcnt)
#define HAVE_TDM_ 1
#endif
#endif

#ifdef HAVE_TDM_
typedef __attribute__((ext_vector_type(4))) unsigned int u32x4_;
typedef __attribute__((ext_vector_type(8))) int          i32x8_;
typedef __attribute__((ext_vector_type(4))) int          i32x4_;
#endif

// ---------------- deterministic hash-normal ----------------
__device__ __forceinline__ uint32_t pcg_(uint32_t x) {
  x = x * 747796405u + 2891336453u;
  uint32_t w = ((x >> ((x >> 28u) + 4u)) ^ x) * 277803737u;
  return (w >> 22u) ^ w;
}
__device__ __forceinline__ float rnormf_(uint32_t s) {
  float a = 0.f;
#pragma unroll
  for (int t = 0; t < 4; ++t) { s = pcg_(s); a += (float)s * 2.3283064e-10f; }
  return (a - 2.0f) * 1.7320508f;   // Irwin-Hall approx N(0,1)
}

// ---------------- block reductions (blockDim.x == 256) ----------------
__device__ __forceinline__ float block_sum_(float v, float* red) {
  int tid = threadIdx.x;
  red[tid] = v; __syncthreads();
  for (int s = 128; s > 0; s >>= 1) { if (tid < s) red[tid] += red[tid + s]; __syncthreads(); }
  float r = red[0]; __syncthreads(); return r;
}
__device__ __forceinline__ float block_max_(float v, float* red) {
  int tid = threadIdx.x;
  red[tid] = v; __syncthreads();
  for (int s = 128; s > 0; s >>= 1) { if (tid < s) red[tid] = fmaxf(red[tid], red[tid + s]); __syncthreads(); }
  float r = red[0]; __syncthreads(); return r;
}

// ---------------- WMMA fragment lane maps (ISA 7.12.2, 16-bit) ----------------
__device__ __forceinline__ int ka_off_(int i, int hi) {
  int v = i >> 1, p = i & 1;
  return (v < 4) ? (hi * 8 + v * 2 + p) : (16 + hi * 8 + (v - 4) * 2 + p);
}
__device__ __forceinline__ int kb_off_(int i, int hi) {
  int v = i >> 1, p = i & 1;
  return (v < 4) ? (hi * 8 + v * 2 + p) : (16 + hi * 8 + (v - 4) * 2 + p);
}

// ================= K1: piece_forward x3 (batched GEMV, b128 streaming) =================
// 64 threads, each owns 4 consecutive output columns -> global_load_b128 rows.
__global__ void k_forward(const float* __restrict__ state,
                          const float* __restrict__ w1, const float* __restrict__ w2,
                          const float* __restrict__ w3,
                          float* __restrict__ pred, float* __restrict__ keysf,
                          float* __restrict__ querf,
                          _Float16* __restrict__ kh, _Float16* __restrict__ qh) {
  __shared__ float xs[D_];
  int bn = blockIdx.x, p = blockIdx.y, tid = threadIdx.x;   // 64 threads
  const float* w  = (p == 0) ? w1 : (p == 1) ? w2 : w3;
  const float* wp = w + (size_t)bn * (D_ * D_);
  {
    float4 v = ((const float4*)(state + bn * D_))[tid];
    xs[tid * 4 + 0] = v.x; xs[tid * 4 + 1] = v.y;
    xs[tid * 4 + 2] = v.z; xs[tid * 4 + 3] = v.w;
  }
  __syncthreads();
  float ax = 0.f, ay = 0.f, az = 0.f, aw = 0.f;
#pragma unroll 4
  for (int j = 0; j < D_; ++j) {
    float4 wr = ((const float4*)(wp + j * D_))[tid];
    float xj = xs[j];
    ax = fmaf(xj, wr.x, ax); ay = fmaf(xj, wr.y, ay);
    az = fmaf(xj, wr.z, az); aw = fmaf(xj, wr.w, aw);
  }
  float4 y;
  y.x = UNTANH_M_ * tanhf(UNTANH_S_ * ax);
  y.y = UNTANH_M_ * tanhf(UNTANH_S_ * ay);
  y.z = UNTANH_M_ * tanhf(UNTANH_S_ * az);
  y.w = UNTANH_M_ * tanhf(UNTANH_S_ * aw);
  int o = bn * D_ + tid * 4;
  if (p == 0) {
    *(float4*)(pred + o) = y;
  } else {
    h4_ hv; hv[0] = (_Float16)y.x; hv[1] = (_Float16)y.y;
    hv[2] = (_Float16)y.z; hv[3] = (_Float16)y.w;
    if (p == 1) { *(float4*)(keysf + o) = y; *(h4_*)(kh + o) = hv; }
    else        { *(float4*)(querf + o) = y; *(h4_*)(qh + o) = hv; }
  }
}

// ================= K2: raw_A = Q.K^T / sqrt(D), WMMA f16 (+TDM staging) =================
__global__ void k_scores(const _Float16* __restrict__ qh, const _Float16* __restrict__ kh,
                         float* __restrict__ rawA) {
  extern __shared__ _Float16 sh2[];
  _Float16* qs = sh2;            // [16][256]
  _Float16* ks = sh2 + 16 * D_;  // [128][256]
  int qt = blockIdx.x, b = blockIdx.y, tid = threadIdx.x;   // 256 threads = 8 waves
  int qbase = (b * N_ + qt * 16) * D_;
  for (int i = tid; i < 16 * D_; i += 256) qs[i] = qh[qbase + i];
  int kbase = b * N_ * D_;
#ifdef HAVE_TDM_
  // Stage K[b] (128 x 256 f16 = 64KB) via Tensor Data Mover, one issuing wave.
  if (tid < 32) {
    uint64_t ga = (uint64_t)(uintptr_t)(kh + kbase);
    u32x4_ g0;
    g0[0] = 1u;                                              // count=1, user mode
    g0[1] = (uint32_t)(uintptr_t)ks;                         // lds_addr (bytes)
    g0[2] = (uint32_t)(ga & 0xffffffffu);                    // global_addr[31:0]
    g0[3] = (uint32_t)((ga >> 32) & 0x01ffffffu) | (2u << 30); // addr[56:32] | type=2
    i32x8_ g1;
    g1[0] = (1 << 16);                    // data_size = 2 bytes
    g1[1] = (int)((D_ & 0xffff) << 16);   // tensor_dim0[15:0]
    g1[2] = (int)((D_ >> 16) | (N_ << 16)); // tensor_dim0[31:16] | tensor_dim1[15:0]
    g1[3] = (int)(D_ << 16);              // tile_dim0
    g1[4] = N_;                           // tile_dim1 (tile_dim2=0)
    g1[5] = D_;                           // tensor_dim0_stride[31:0]
    g1[6] = 0;
    g1[7] = 0;
    i32x4_ gz = {0, 0, 0, 0};
#if __clang_major__ >= 23
    i32x8_ gz8 = {0, 0, 0, 0, 0, 0, 0, 0};
    __builtin_amdgcn_tensor_load_to_lds(g0, g1, gz, gz, gz8, 0);
#else
    __builtin_amdgcn_tensor_load_to_lds(g0, g1, gz, gz, 0);
#endif
    __builtin_amdgcn_s_wait_tensorcnt(0);
  }
#else
  for (int i = tid; i < N_ * D_; i += 256) ks[i] = kh[kbase + i];
#endif
  __syncthreads();
  int lane = tid & 31, wave = tid >> 5;        // wave = k-tile (16 cols each)
  int hi = lane >> 4, ln = lane & 15;
  v8f acc = {};
#pragma unroll
  for (int c = 0; c < D_ / 32; ++c) {
    v16h a = {}, bf = {};
#pragma unroll
    for (int i = 0; i < 16; ++i) {
      a[i]  = qs[ln * D_ + c * 32 + ka_off_(i, hi)];
      bf[i] = ks[(wave * 16 + ln) * D_ + c * 32 + kb_off_(i, hi)];
    }
    acc = __builtin_amdgcn_wmma_f32_16x16x32_f16(false, a, false, bf, (short)0, acc, false, false);
  }
#pragma unroll
  for (int r = 0; r < 8; ++r) {
    int row = qt * 16 + r + 8 * hi;
    int col = wave * 16 + ln;
    rawA[((size_t)b * N_ + row) * N_ + col] = acc[r] * 0.0625f;   // 1/sqrt(256)
  }
}

// ================= K3: rms_norm(-1,-2) + EMA + Sinkhorn + threshold + row var =================
__global__ void k_sinkhorn(const float* __restrict__ rawA, const float* __restrict__ A_ema,
                           const int* __restrict__ stepc,
                           float* __restrict__ Aout, _Float16* __restrict__ Ah,
                           float* __restrict__ vd) {
  extern __shared__ float s3[];
  float* P   = s3;            // 16384
  float* rs  = s3 + 16384;    // 128
  float* cs  = s3 + 16512;    // 128
  float* red = s3 + 16640;    // 256
  int b = blockIdx.x, tid = threadIdx.x;
  const float* ra = rawA  + (size_t)b * N_ * N_;
  const float* ae = A_ema + (size_t)b * N_ * N_;
  float acc = 0.f;
  for (int i = tid; i < N_ * N_; i += 256) { float x = ra[i]; P[i] = x; acc += x * x; }
  float ssq = block_sum_(acc, red);
  float scale = rsqrtf(ssq / (float)(N_ * N_) + EPS_);
  int step = stepc[0];
  for (int i = tid; i < N_ * N_; i += 256) {
    float a  = P[i] * scale;
    float an = (step == 1) ? a : ae[i] * C_MOMENT_ + a * (1.f - C_MOMENT_);
    P[i] = expf(an);
  }
  __syncthreads();
  for (int it = 0; it < 10; ++it) {
    if (tid < N_) { float s = 0.f; for (int j = 0; j < N_; ++j) s += P[tid * N_ + j]; rs[tid] = s + EPS_; }
    __syncthreads();
    for (int i = tid; i < N_ * N_; i += 256) P[i] /= rs[i >> 7];
    __syncthreads();
    if (tid < N_) { float s = 0.f; for (int r = 0; r < N_; ++r) s += P[r * N_ + tid]; cs[tid] = s + EPS_; }
    __syncthreads();
    for (int i = tid; i < N_ * N_; i += 256) P[i] /= cs[i & (N_ - 1)];
    __syncthreads();
  }
  const float ub = 1.f / (N_ + 1.f);
  for (int i = tid; i < N_ * N_; i += 256) {
    float v = P[i];
    v = (v > ub) ? v : 0.f;
    if ((i >> 7) < 2) v = 0.f;          // rows 0,1 zeroed
    P[i] = v;
    Aout[(size_t)b * N_ * N_ + i] = v;
    Ah[(size_t)b * N_ * N_ + i]   = (_Float16)v;
  }
  __syncthreads();
  if (tid < N_) {
    float m = 0.f, m2 = 0.f;
    for (int j = 0; j < N_; ++j) { float v = P[tid * N_ + j]; m += v; m2 += v * v; }
    m /= (float)N_; m2 /= (float)N_;
    float d = A_VAR_ - (m2 - m * m);
    vd[b * N_ + tid] = d > 0.f ? d : 0.f;
  }
}

// ================= K4: target = A x (output + vd*noise), WMMA f16; env rows pinned =================
__global__ void k_target(const _Float16* __restrict__ Ah, const float* __restrict__ output,
                         const float* __restrict__ vd,
                         const float* __restrict__ eye, const float* __restrict__ stomach,
                         float* __restrict__ tgt) {
  extern __shared__ _Float16 sh4[];
  _Float16* As = sh4;            // [16][128]
  _Float16* Vs = sh4 + 16 * N_;  // [128][256]
  int nt = blockIdx.x, b = blockIdx.y, tid = threadIdx.x;   // 512 threads = 16 waves
  int abase = (b * N_ + nt * 16) * N_;
  for (int i = tid; i < 16 * N_; i += 512) As[i] = Ah[abase + i];
  for (int i = tid; i < N_ * D_; i += 512) {
    int m = i >> 8;
    float nz = vd[b * N_ + m] * rnormf_(0x51ED0000u ^ (uint32_t)(b * N_ * D_ + i));
    Vs[i] = (_Float16)(output[(size_t)b * N_ * D_ + i] + nz);
  }
  __syncthreads();
  int lane = tid & 31, wave = tid >> 5;   // wave = d-tile (16 cols each)
  int hi = lane >> 4, ln = lane & 15;
  v8f acc = {};
#pragma unroll
  for (int c = 0; c < N_ / 32; ++c) {
    v16h a = {}, bf = {};
#pragma unroll
    for (int i = 0; i < 16; ++i) {
      a[i]  = As[ln * N_ + c * 32 + ka_off_(i, hi)];
      bf[i] = Vs[(c * 32 + kb_off_(i, hi)) * D_ + wave * 16 + ln];
    }
    acc = __builtin_amdgcn_wmma_f32_16x16x32_f16(false, a, false, bf, (short)0, acc, false, false);
  }
#pragma unroll
  for (int r = 0; r < 8; ++r) {
    int n = nt * 16 + r + 8 * hi;
    int d = wave * 16 + ln;
    float v = acc[r];
    if (n == 0)      v = eye[b * D_ + d];
    else if (n == 1) v = stomach[b * D_ + d];
    tgt[((size_t)b * N_ + n) * D_ + d] = v;
  }
}

// ================= K5: errors + softmax/eb/plasticity -> rank-1 grad factors =================
__global__ void k_grads(const float* __restrict__ pred, const float* __restrict__ tgt,
                        const float* __restrict__ keysf, const float* __restrict__ querf,
                        const float* __restrict__ state,
                        const float* __restrict__ eb1, const float* __restrict__ eb2,
                        const float* __restrict__ eb3,
                        float* __restrict__ rowc, float* __restrict__ noisyb) {
  __shared__ float red[256];
  int bn = blockIdx.x, tid = threadIdx.x;
  int idx = bn * D_ + tid;
  float pr = pred[idx], tg = tgt[idx], ky = keysf[idx], qr = querf[idx];
  float e1b = pr - tg;
  float e2  = ky - e1b;
  float e3  = qr - e2;
  float e1  = pr - tg - e3;
  float st  = state[idx];
  float errs[3] = { e1, e2, e3 };
  const float* ebp[3] = { eb1, eb2, eb3 };
  for (int p = 0; p < 3; ++p) {
    float err = errs[p];
    float mx  = block_max_(err, red);
    float ex  = expf(err - mx);
    float sm  = block_sum_(ex, red);
    float E   = ex / sm;
    float eb  = ebp[p][idx];
    float ebn = eb * (1.f - EMA_SPEED_) + EMA_SPEED_ * E;
    float adv = E - ebn;
    float plast = 1.f + adv * rsqrtf(adv * adv + EPS_);   // rms_norm over size-1 axis
    float noisy = st + NOISE_SCALE_ * rnormf_(0x9E370000u + (uint32_t)p * 0x1234567u + (uint32_t)idx);
    float msq   = block_sum_(noisy * noisy, red) * (1.f / (float)D_);
    float rc    = -plast * err * rsqrtf(err * err * msq + EPS_);
    rowc[(size_t)p * BND_ + idx]   = rc;     // grad[i,j] = rowc[i] * noisy[j]
    noisyb[(size_t)p * BND_ + idx] = noisy;
  }
}

// ================= K6: fused piece_step; w_tmp in 256KB LDS; b128 streaming =================
// 256 threads; per iteration each thread handles float4 of one of 4 rows.
__global__ void k_update(const float* __restrict__ w1, const float* __restrict__ w2,
                         const float* __restrict__ w3,
                         const float* __restrict__ g1, const float* __restrict__ g2,
                         const float* __restrict__ g3,
                         const float* __restrict__ rowc, const float* __restrict__ noisyb,
                         const int* __restrict__ stepc,
                         float* __restrict__ wout) {
  extern __shared__ float s6[];
  float* wt  = s6;            // 65536 floats = 256KB (w_tmp tile)
  float* red = s6 + 65536;    // 256
  int bn = blockIdx.x, p = blockIdx.y, tid = threadIdx.x;
  const float* w  = (p == 0) ? w1 : (p == 1) ? w2 : w3;
  const float* ge = (p == 0) ? g1 : (p == 1) ? g2 : g3;
  const float* wp = w  + (size_t)bn * D_ * D_;
  const float* gp = ge + (size_t)bn * D_ * D_;
  const float* rc = rowc   + (size_t)p * BND_ + bn * D_;
  const float* ns = noisyb + (size_t)p * BND_ + bn * D_;
  float* op = wout + (size_t)p * WSZ_ + (size_t)bn * D_ * D_;

  int cg = (tid & 63) * 4;      // column group (4 consecutive j)
  int ro = tid >> 6;            // row offset 0..3
  float step   = (float)stepc[0];
  float center = fmodf(step * SPEED_, (float)D_);
  float sg = WIDTH_ * (float)D_;
  float mk[4], nj[4];
  {
    float4 n4 = *(const float4*)(ns + cg);
    nj[0] = n4.x; nj[1] = n4.y; nj[2] = n4.z; nj[3] = n4.w;
#pragma unroll
    for (int c = 0; c < 4; ++c) {
      float dj = fabsf((float)(cg + c) - center);
      dj = fminf(dj, (float)D_ - dj);
      mk[c] = expf(-0.5f * (dj / sg) * (dj / sg));
    }
  }
  float acc = 0.f;
  for (int it = 0; it < D_ / 4; ++it) {
    int i = it * 4 + ro;                       // row (err index)
    size_t off = (size_t)i * D_ + cg;
    float4 wv = *(const float4*)(wp + off);
    float4 gv = *(const float4*)(gp + off);
    float rcv = rc[i];
    float4 v;
    v.x = fmaf(LR_, MOMENTUM_ * gv.x + (1.f - MOMENTUM_) * (rcv * nj[0] - WD_ * wv.x) * mk[0], wv.x);
    v.y = fmaf(LR_, MOMENTUM_ * gv.y + (1.f - MOMENTUM_) * (rcv * nj[1] - WD_ * wv.y) * mk[1], wv.y);
    v.z = fmaf(LR_, MOMENTUM_ * gv.z + (1.f - MOMENTUM_) * (rcv * nj[2] - WD_ * wv.z) * mk[2], wv.z);
    v.w = fmaf(LR_, MOMENTUM_ * gv.w + (1.f - MOMENTUM_) * (rcv * nj[3] - WD_ * wv.w) * mk[3], wv.w);
    *(float4*)(wt + off) = v;                  // ds_store_b128
    acc += v.x * v.x + v.y * v.y + v.z * v.z + v.w * v.w;
  }
  red[tid] = acc; __syncthreads();
  for (int s = 128; s > 0; s >>= 1) { if (tid < s) red[tid] += red[tid + s]; __syncthreads(); }
  float scale = W_SCALE_ * rsqrtf(red[0] / (float)(D_ * D_) + EPS_);
  __syncthreads();
  for (int it = 0; it < D_ / 4; ++it) {
    int i = it * 4 + ro;
    size_t off = (size_t)i * D_ + cg;
    float4 v = *(const float4*)(wt + off);     // ds_load_b128
    v.x *= scale; v.y *= scale; v.z *= scale; v.w *= scale;
    *(float4*)(op + off) = v;                  // global_store_b128
  }
}

// ================================= launcher =================================
extern "C" void kernel_launch(void* const* d_in, const int* in_sizes, int n_in,
                              void* d_out, int out_size, void* d_ws, size_t ws_size,
                              hipStream_t stream) {
  (void)in_sizes; (void)n_in; (void)out_size; (void)ws_size;
  const float* state   = (const float*)d_in[0];
  const float* output  = (const float*)d_in[1];
  const float* A_ema   = (const float*)d_in[2];
  const float* w1 = (const float*)d_in[3];
  const float* w2 = (const float*)d_in[4];
  const float* w3 = (const float*)d_in[5];
  const float* eb1 = (const float*)d_in[6];
  const float* eb2 = (const float*)d_in[7];
  const float* eb3 = (const float*)d_in[8];
  const float* g1 = (const float*)d_in[9];
  const float* g2 = (const float*)d_in[10];
  const float* g3 = (const float*)d_in[11];
  const float* eye     = (const float*)d_in[12];
  const float* stomach = (const float*)d_in[13];
  const int*   stepc   = (const int*)d_in[14];

  float* out  = (float*)d_out;
  float* pred = out;                        // [B,N,D]
  float* tgt  = out + BND_;                 // [B,N,D]
  float* Aout = out + 2 * BND_;             // [B,N,N]
  float* wout = out + 2 * BND_ + BNN_;      // w1n|w2n|w3n contiguous

  float* ws     = (float*)d_ws;
  float* rawA   = ws;                       // 65536
  float* vd     = ws + 65536;               // 512
  float* keysf  = ws + 66048;               // 131072
  float* querf  = ws + 197120;              // 131072
  float* rowc   = ws + 328192;              // 3*131072
  float* noisyb = ws + 721408;              // 3*131072
  _Float16* qh = (_Float16*)(ws + 1114624); // 131072 halves
  _Float16* kh = qh + BND_;                 // 131072 halves
  _Float16* Ah = kh + BND_;                 // 65536 halves

  k_forward<<<dim3(BN_, 3), 64, 0, stream>>>(state, w1, w2, w3, pred, keysf, querf, kh, qh);
  k_scores<<<dim3(8, B_), 256, (16 * D_ + N_ * D_) * sizeof(_Float16), stream>>>(qh, kh, rawA);
  k_sinkhorn<<<B_, 256, (16384 + 128 + 128 + 256) * sizeof(float), stream>>>(rawA, A_ema, stepc, Aout, Ah, vd);
  k_target<<<dim3(8, B_), 512, (16 * N_ + N_ * D_) * sizeof(_Float16), stream>>>(Ah, output, vd, eye, stomach, tgt);
  k_grads<<<BN_, D_, 0, stream>>>(pred, tgt, keysf, querf, state, eb1, eb2, eb3, rowc, noisyb);
  k_update<<<dim3(BN_, 3), D_, (65536 + 256) * sizeof(float), stream>>>(w1, w2, w3, g1, g2, g3, rowc, noisyb, stepc, wout);
}
